// ListWiseReformer_41051297415809
// MI455X (gfx1250) — matrix operand order repaired and checked by reading, not compiled, not run on hardware
//
#include <hip/hip_runtime.h>

typedef __attribute__((ext_vector_type(16))) __bf16 v16bf;
typedef __attribute__((ext_vector_type(8)))  float  v8f;
typedef __attribute__((ext_vector_type(4)))  unsigned int v4u;
typedef __attribute__((ext_vector_type(8)))  int v8i;
typedef __attribute__((ext_vector_type(4)))  int v4i;

#define SEQ    1024
#define DIM    768
#define FDIM   3072
#define NHEAD  8
#define DHEAD  96
#define NBATCH 16
#define NBH    128           // NBATCH*NHEAD
#define NHASH  4
#define NSORT  4096          // NHASH*SEQ
#define NCHUNK 64            // NHASH * (SEQ/64)
#define NROWS  16384         // NBATCH*SEQ
#define DEPTH  6

__device__ __forceinline__ __bf16 f2bf(float x) {
  unsigned u = __float_as_uint(x);
  u = u + 0x7FFFu + ((u >> 16) & 1u);          // round-to-nearest-even
  unsigned short h = (unsigned short)(u >> 16);
  __bf16 r; __builtin_memcpy(&r, &h, 2); return r;
}

// ---------------------------------------------------------------------------
// Tensor Data Mover: 2D tile load Global -> LDS.
// Descriptor per CDNA5 ISA ch.8 (D# group0/group1).
//   data_size=2B; tile = tile_rows x tile_k elems; tensor row length = rowlen;
//   LDS padding: +4 DWORDs (16B) after every 16 DWORDs (64B) => 40-elem rows.
// ---------------------------------------------------------------------------
__device__ __forceinline__ void tdm_load_tile_2d(const __bf16* gsrc, void* ldst,
                                                 int rowlen, int nrows,
                                                 int tile_k, int tile_rows)
{
  unsigned long long ga = (unsigned long long)(uintptr_t)gsrc;
  unsigned lo = (unsigned)(uintptr_t)ldst;     // low 32 bits of flat shared ptr = LDS offset
  v4u g0;
  g0[0] = 1u;                                  // count=1, user descriptor
  g0[1] = lo;                                  // lds_addr
  g0[2] = (unsigned)ga;                        // global_addr[31:0]
  g0[3] = ((unsigned)(ga >> 32) & 0x01FFFFFFu) | (2u << 30);  // ga[56:32] | type=2
  v8i g1;
  g1[0] = (int)((1u << 16)                     // data_size = 2 bytes
              | (1u << 20)                     // pad_enable
              | (3u << 22)                     // pad_interval: 16 DWORDs
              | (3u << 25));                   // pad_amount:   4 DWORDs
  g1[1] = (int)(((unsigned)rowlen & 0xFFFFu) << 16);                    // dim0[15:0]
  g1[2] = (int)(((unsigned)rowlen >> 16) | (((unsigned)nrows) << 16));  // dim0[31:16]|dim1[15:0]
  g1[3] = (int)(((unsigned)tile_k) << 16);                              // dim1[31:16]=0 | tile_dim0
  g1[4] = (int)((unsigned)tile_rows & 0xFFFFu);                         // tile_dim1 (tile_dim2=0)
  g1[5] = (int)rowlen;                                                  // dim0_stride[31:0]
  g1[6] = 0;                                                            // stride hi / dim1_stride lo
  g1[7] = 0;
  v4i z4 = {0, 0, 0, 0};
#if __has_include(<hip/amd_detail/amd_gfx1250_TDM.h>)
  v8i z8 = {0, 0, 0, 0, 0, 0, 0, 0};
  __builtin_amdgcn_tensor_load_to_lds(g0, g1, z4, z4, z8, 0);   // clang-23 / therock arity
#else
  __builtin_amdgcn_tensor_load_to_lds(g0, g1, z4, z4, 0);       // ROCm 7.2 clang-22 arity
#endif
}

// ---------------------------------------------------------------------------
// bf16 WMMA GEMM with TDM-fed, double-buffered LDS tiles.
// A: MxK row-major bf16.  Bt: NxK row-major bf16 (pre-transposed weights).
// mode 0: fp32 row-major   mode 1: fp32 head-major (B,T,H,dh)->(BH,T,dh)
// mode 2: fp32 resid + acc + bias   mode 3: bf16 gelu(acc + bias)
// ---------------------------------------------------------------------------
#define GM 128
#define GN 128
#define GK 32
#define LDT 40   // padded LDS k-stride (bf16): 64B data + 16B TDM pad per row

__global__ __launch_bounds__(256)
void gemm_bf16_wmma_kernel(const __bf16* __restrict__ A, const __bf16* __restrict__ Bt,
                           const float* __restrict__ bias, const float* __restrict__ resid,
                           void* __restrict__ outp, int M, int N, int K, int mode)
{
  __shared__ __align__(16) __bf16 sA[2][GM * LDT];
  __shared__ __align__(16) __bf16 sB[2][GN * LDT];
  const int tid  = threadIdx.x;
  const int bm   = blockIdx.x * GM;
  const int bn   = blockIdx.y * GN;
  const int lane = tid & 31;
  const int wave = tid >> 5;
  const int wm   = (wave & 3) * 32;   // 2 M-tiles per wave
  const int wn   = (wave >> 2) * 64;  // 4 N-tiles per wave

  const __bf16* Ablk = A  + (size_t)bm * K;
  const __bf16* Bblk = Bt + (size_t)bn * K;
  if (mode == 2) __builtin_prefetch(resid + (size_t)bm * N + bn, 0, 1);

  const v8f vz = {0.f,0.f,0.f,0.f,0.f,0.f,0.f,0.f};
  v8f acc[2][4];
#pragma unroll
  for (int r = 0; r < 2; r++)
#pragma unroll
    for (int c2 = 0; c2 < 4; c2++) acc[r][c2] = vz;

  const int nsteps = K / GK;
  if (wave == 0) {                       // TDM issue: once per wave, EXEC-independent
    tdm_load_tile_2d(Ablk, &sA[0][0], K, GM, GK, GM);
    tdm_load_tile_2d(Bblk, &sB[0][0], K, GN, GK, GN);
  }

  int buf = 0;
  for (int t = 0; t < nsteps; ++t) {
    if (wave == 0) __builtin_amdgcn_s_wait_tensorcnt(0);  // tiles for step t arrived
    __syncthreads();                                      // publish to all waves
    if (wave == 0 && (t + 1) < nsteps) {                  // prefetch step t+1 via TDM
      tdm_load_tile_2d(Ablk + (size_t)(t + 1) * GK, &sA[buf ^ 1][0], K, GM, GK, GM);
      tdm_load_tile_2d(Bblk + (size_t)(t + 1) * GK, &sB[buf ^ 1][0], K, GN, GK, GN);
    }

    // Fragments per CDNA5 16-bit layouts (ISA 7.12.2)
    v16bf af[2], bfrag[4];
    const int arow = wm + (lane & 15);
    const int akof = (lane >> 4) * 8;
#pragma unroll
    for (int r = 0; r < 2; r++) {
      const __bf16* p = &sA[buf][(arow + r * 16) * LDT + akof];
      ((uint4*)&af[r])[0] = *(const uint4*)p;        // K = akof .. akof+7
      ((uint4*)&af[r])[1] = *(const uint4*)(p + 16); // K = akof+16 .. akof+23
    }
    const int bcol = wn + (lane & 15);
    const int bkof = (lane >> 4) * 8;
#pragma unroll
    for (int c2 = 0; c2 < 4; c2++) {
      const __bf16* p = &sB[buf][(bcol + c2 * 16) * LDT + bkof];
      ((uint4*)&bfrag[c2])[0] = *(const uint4*)p;        // K = bkof .. bkof+7
      ((uint4*)&bfrag[c2])[1] = *(const uint4*)(p + 16); // K = bkof+16 .. bkof+23
    }
#pragma unroll
    for (int r = 0; r < 2; r++)
#pragma unroll
      for (int c2 = 0; c2 < 4; c2++)
        acc[r][c2] = __builtin_amdgcn_wmma_f32_16x16x32_bf16(
            false, af[r], false, bfrag[c2], (short)0, acc[r][c2], false, false);
    buf ^= 1;
  }

  float*  outf = (float*)outp;
  __bf16* outb = (__bf16*)outp;
#pragma unroll
  for (int r = 0; r < 2; r++) {
#pragma unroll
    for (int c2 = 0; c2 < 4; c2++) {
#pragma unroll
      for (int j = 0; j < 8; j++) {
        int m = bm + wm + r * 16 + ((lane >> 4) << 3) + j;
        int n = bn + wn + c2 * 16 + (lane & 15);
        float v = acc[r][c2][j];
        if (mode == 0) {
          outf[(size_t)m * N + n] = v;
        } else if (mode == 1) {
          int b = m >> 10, t = m & (SEQ - 1);
          int h = n / DHEAD, e = n - h * DHEAD;
          outf[(((size_t)(b * NHEAD + h)) * SEQ + t) * DHEAD + e] = v;
        } else if (mode == 2) {
          size_t o = (size_t)m * N + n;
          outf[o] = resid[o] + v + bias[n];
        } else {
          float s = v + bias[n];
          outb[(size_t)m * N + n] = f2bf(0.5f * s * (1.0f + erff(s * 0.70710678f)));
        }
      }
    }
  }
  (void)M;
}

// ---------------------------------------------------------------------------
// Weight prep: fp32 (L,K,N) -> bf16 (L,N,K)  (tiled LDS transpose)
// ---------------------------------------------------------------------------
__global__ __launch_bounds__(256)
void transpose_bf16_kernel(const float* __restrict__ src, __bf16* __restrict__ dst,
                           int K, int N) {
  __shared__ float tile[32][33];
  const int l = blockIdx.z;
  src += (size_t)l * K * N;
  dst += (size_t)l * K * N;
  const int k0 = blockIdx.x * 32, n0 = blockIdx.y * 32;
  const int tn = threadIdx.x, tk = threadIdx.y;   // 32 x 8
#pragma unroll
  for (int i = 0; i < 32; i += 8)
    tile[tk + i][tn] = src[(size_t)(k0 + tk + i) * N + n0 + tn];
  __syncthreads();
#pragma unroll
  for (int i = 0; i < 32; i += 8)
    dst[(size_t)(n0 + tk + i) * K + k0 + tn] = f2bf(tile[tn][tk + i]);
}

// ---------------------------------------------------------------------------
// Elementwise helpers
// ---------------------------------------------------------------------------
__global__ __launch_bounds__(256)
void embed_kernel(const int* __restrict__ x, const float* __restrict__ tok,
                  const float* __restrict__ pos, float* __restrict__ x1,
                  float* __restrict__ x2) {
  size_t idx = (size_t)blockIdx.x * 256 + threadIdx.x;  // NROWS*DIM total
  int row = (int)(idx / DIM);
  int cc  = (int)(idx % DIM);
  int t   = row & (SEQ - 1);
  int tk  = x[row];
  float h = tok[(size_t)tk * DIM + cc] + pos[(size_t)t * DIM + cc];
  x1[idx] = h; x2[idx] = h;
}

__global__ __launch_bounds__(256)
void layernorm_bf16_kernel(const float* __restrict__ x, const float* __restrict__ g,
                           const float* __restrict__ bta, __bf16* __restrict__ out) {
  int row = blockIdx.x;
  const float* xr = x + (size_t)row * DIM;
  float s = 0.f, sq = 0.f;
  for (int c = threadIdx.x; c < DIM; c += 256) { float v = xr[c]; s += v; sq += v * v; }
  for (int m = 16; m > 0; m >>= 1) { s += __shfl_xor(s, m); sq += __shfl_xor(sq, m); }
  __shared__ float rs[8], rq[8];
  int w = threadIdx.x >> 5, ln = threadIdx.x & 31;
  if (ln == 0) { rs[w] = s; rq[w] = sq; }
  __syncthreads();
  if (w == 0) {
    s  = (ln < 8) ? rs[ln] : 0.f;
    sq = (ln < 8) ? rq[ln] : 0.f;
    for (int m = 4; m > 0; m >>= 1) { s += __shfl_xor(s, m); sq += __shfl_xor(sq, m); }
    if (ln == 0) { rs[0] = s; rq[0] = sq; }
  }
  __syncthreads();
  float mu  = rs[0] * (1.0f / DIM);
  float var = rq[0] * (1.0f / DIM) - mu * mu;
  float inv = rsqrtf(var + 1e-5f);
  __bf16* orow = out + (size_t)row * DIM;
  for (int c = threadIdx.x; c < DIM; c += 256)
    orow[c] = f2bf((xr[c] - mu) * inv * g[c] + bta[c]);
}

// ---------------------------------------------------------------------------
// LSH pipeline
// ---------------------------------------------------------------------------
__global__ __launch_bounds__(256)
void rotgen_kernel(float* __restrict__ rot, int layer) {
  int i = blockIdx.x * 256 + threadIdx.x;
  if (i < DHEAD * 32) {
    unsigned s = (unsigned)(layer + 1) * 0x9E3779B9u ^ (unsigned)i * 0x85EBCA6Bu;
    s ^= s >> 16; s *= 0x7FEB352Du; s ^= s >> 15; s *= 0x846CA68Bu; s ^= s >> 16;
    unsigned s2 = s + 0x9E3779B9u;
    s2 ^= s2 >> 16; s2 *= 0x7FEB352Du; s2 ^= s2 >> 15; s2 *= 0x846CA68Bu; s2 ^= s2 >> 16;
    float u1 = ((float)s + 0.5f) * (1.0f / 4294967296.0f);
    float u2 = ((float)s2) * (1.0f / 4294967296.0f);
    rot[i] = sqrtf(-2.0f * __logf(u1)) * __cosf(6.2831853f * u2);
  }
}

__global__ __launch_bounds__(256)
void lsh_hash_kernel(const float* __restrict__ qk, const float* __restrict__ rot,
                     int* __restrict__ buckets) {
  int idx = blockIdx.x * 256 + threadIdx.x;  // NBH*SEQ
  int bh = idx >> 10, t = idx & (SEQ - 1);
  const float* q = qk + ((size_t)bh * SEQ + t) * DHEAD;
  float dots[32];
#pragma unroll
  for (int i = 0; i < 32; i++) dots[i] = 0.f;
  for (int f = 0; f < DHEAD; f++) {
    float qf = q[f];
    const float* rf = rot + f * 32;
#pragma unroll
    for (int i = 0; i < 32; i++) dots[i] += qf * rf[i];
  }
#pragma unroll
  for (int h = 0; h < NHASH; h++) {
    float best = -3.4e38f; int bi = 0;
#pragma unroll
    for (int i = 0; i < 8; i++) {
      float d = dots[h * 8 + i];
      if (d  > best) { best = d;  bi = i;     }
      if (-d > best) { best = -d; bi = i + 8; }
    }
    buckets[(size_t)bh * NSORT + h * SEQ + t] = bi + h * 16;
  }
}

__global__ __launch_bounds__(1024)
void lsh_sort_kernel(const int* __restrict__ buckets, int* __restrict__ sticker,
                     int* __restrict__ undo) {
  int bh = blockIdx.x;
  __shared__ int sk_[NSORT];
  __shared__ int sv_[NSORT];
  for (int e = threadIdx.x; e < NSORT; e += 1024) {
    sk_[e] = buckets[(size_t)bh * NSORT + e] * SEQ + (e & (SEQ - 1));
    sv_[e] = e;
  }
  __syncthreads();
  for (int len = 2; len <= NSORT; len <<= 1) {
    for (int st2 = len >> 1; st2 > 0; st2 >>= 1) {
      for (int i = threadIdx.x; i < NSORT / 2; i += 1024) {
        int a  = ((i & ~(st2 - 1)) << 1) | (i & (st2 - 1));
        int b2 = a + st2;
        bool up = ((a & len) == 0);
        int ka = sk_[a], kb = sk_[b2];
        if ((ka > kb) == up) {
          sk_[a] = kb; sk_[b2] = ka;
          int va = sv_[a]; sv_[a] = sv_[b2]; sv_[b2] = va;
        }
      }
      __syncthreads();
    }
  }
  for (int e = threadIdx.x; e < NSORT; e += 1024) {
    sticker[(size_t)bh * NSORT + e] = sv_[e];
    undo[(size_t)bh * NSORT + sv_[e]] = e;
  }
}

// Per (bh, chunk): dots = q(64x96) @ k^T(96x128) via WMMA, in-register softmax,
// bo = p(64x128) @ v(128x96) via WMMA.  4 waves, M-strip of 16 rows per wave.
__global__ __launch_bounds__(128)
void lsh_attn_kernel(const float* __restrict__ qk, const float* __restrict__ vv,
                     const int* __restrict__ sticker, float* __restrict__ so,
                     float* __restrict__ slog) {
  __shared__ __align__(16) char smem[24576 + 16384 + 64 * 4 + 128 * 4];
  __bf16* s_k  = (__bf16*)smem;                    // [128][96] stage1; reused as s_vt[96][128]
  __bf16* s_q  = (__bf16*)(smem + 24576);          // [64][96]  stage1; reused as s_p[64][128]
  int*    s_qt = (int*)(smem + 24576 + 16384);     // [64]
  int*    s_kt = s_qt + 64;                        // [128]

  const int tid  = threadIdx.x;
  const int c    = blockIdx.x;   // chunk
  const int bh   = blockIdx.y;
  const int lane = tid & 31;
  const int wave = tid >> 5;
  const int* stk = sticker + (size_t)bh * NSORT;

  // gather K rows (look_back: [chunk c, chunk c-1]) with L2-normalization
  int j    = tid;                               // 0..127, one key row each
  int cj   = (j < 64) ? c : ((c + NCHUNK - 1) & (NCHUNK - 1));
  int kpos = stk[cj * 64 + (j & 63)] & (SEQ - 1);
  s_kt[j] = kpos;
  {
    const float* kr = qk + ((size_t)bh * SEQ + kpos) * DHEAD;
    float nrm = 0.f;
    for (int e = 0; e < DHEAD; e++) { float x = kr[e]; nrm += x * x; }
    float inv = 1.f / fmaxf(sqrtf(nrm), 1e-6f);
    for (int e = 0; e < DHEAD; e++) s_k[j * DHEAD + e] = f2bf(kr[e] * inv);
  }
  if (tid < 64) {
    int qpos = stk[c * 64 + tid] & (SEQ - 1);
    s_qt[tid] = qpos;
    const float* qr = qk + ((size_t)bh * SEQ + qpos) * DHEAD;
    for (int e = 0; e < DHEAD; e++) s_q[tid * DHEAD + e] = f2bf(qr[e]);
  }
  __syncthreads();

  // stage 1: dots
  const v8f vz = {0.f,0.f,0.f,0.f,0.f,0.f,0.f,0.f};
  v8f acc[8];
#pragma unroll
  for (int nt = 0; nt < 8; nt++) acc[nt] = vz;
  const int mrow = wave * 16;
#pragma unroll
  for (int ks = 0; ks < 3; ks++) {
    v16bf aq;
    int ar = mrow + (lane & 15);
    int ak = ((lane >> 4) << 3) + ks * 32;
    const __bf16* pa = &s_q[ar * DHEAD + ak];
    ((uint4*)&aq)[0] = *(const uint4*)pa;
    ((uint4*)&aq)[1] = *(const uint4*)(pa + 16);
#pragma unroll
    for (int nt = 0; nt < 8; nt++) {
      v16bf bk;
      int bc  = nt * 16 + (lane & 15);
      int bkk = ((lane >> 4) << 3) + ks * 32;
      const __bf16* pb = &s_k[bc * DHEAD + bkk];
      ((uint4*)&bk)[0] = *(const uint4*)pb;
      ((uint4*)&bk)[1] = *(const uint4*)(pb + 16);
      acc[nt] = __builtin_amdgcn_wmma_f32_16x16x32_bf16(
          false, aq, false, bk, (short)0, acc[nt], false, false);
    }
  }

  // softmax in registers: rows live across 16-lane half-wave (wave32 C layout)
  const int rbase = mrow + ((lane >> 4) << 3);
  const int ncol  = lane & 15;
#pragma unroll
  for (int j2 = 0; j2 < 8; j2++) {
    int row = rbase + j2;
    int myt = s_qt[row];
    float mx = -3.0e38f;
    float dv[8];
#pragma unroll
    for (int nt = 0; nt < 8; nt++) {
      float d2 = acc[nt][j2] * 0.1020620726f;   // 1/sqrt(96)
      if (s_kt[nt * 16 + ncol] == myt) d2 = -5.0e4f;  // self-attn mask
      dv[nt] = d2;
      mx = fmaxf(mx, d2);
    }
    mx = fmaxf(mx, __shfl_xor(mx, 1));
    mx = fmaxf(mx, __shfl_xor(mx, 2));
    mx = fmaxf(mx, __shfl_xor(mx, 4));
    mx = fmaxf(mx, __shfl_xor(mx, 8));
    float sum = 0.f;
#pragma unroll
    for (int nt = 0; nt < 8; nt++) { dv[nt] = __expf(dv[nt] - mx); sum += dv[nt]; }
    sum += __shfl_xor(sum, 1);
    sum += __shfl_xor(sum, 2);
    sum += __shfl_xor(sum, 4);
    sum += __shfl_xor(sum, 8);
    float inv = 1.f / sum;
#pragma unroll
    for (int nt = 0; nt < 8; nt++) acc[nt][j2] = dv[nt] * inv;
    if (ncol == 0) slog[(size_t)bh * NSORT + c * 64 + row] = mx + __logf(sum);
  }
  __syncthreads();  // everyone done reading s_q/s_k

  __bf16* s_p  = s_q;   // reuse: [64][128]
  __bf16* s_vt = s_k;   // reuse: [96][128] (transposed V)
#pragma unroll
  for (int nt = 0; nt < 8; nt++)
#pragma unroll
    for (int j2 = 0; j2 < 8; j2++)
      s_p[(rbase + j2) * 128 + nt * 16 + ncol] = f2bf(acc[nt][j2]);
  {
    const float* vr = vv + ((size_t)bh * SEQ + kpos) * DHEAD;
    for (int e = 0; e < DHEAD; e++) s_vt[e * 128 + j] = f2bf(vr[e]);
  }
  __syncthreads();

  // stage 2: bo = p @ v
  v8f acc2[6];
#pragma unroll
  for (int nt = 0; nt < 6; nt++) acc2[nt] = vz;
#pragma unroll
  for (int ks = 0; ks < 4; ks++) {
    v16bf ap;
    int ar = mrow + (lane & 15);
    int ak = ((lane >> 4) << 3) + ks * 32;
    const __bf16* pa = &s_p[ar * 128 + ak];
    ((uint4*)&ap)[0] = *(const uint4*)pa;
    ((uint4*)&ap)[1] = *(const uint4*)(pa + 16);
#pragma unroll
    for (int nt = 0; nt < 6; nt++) {
      v16bf bvf;
      int bc  = nt * 16 + (lane & 15);
      int bkk = ((lane >> 4) << 3) + ks * 32;
      const __bf16* pb = &s_vt[bc * 128 + bkk];
      ((uint4*)&bvf)[0] = *(const uint4*)pb;
      ((uint4*)&bvf)[1] = *(const uint4*)(pb + 16);
      acc2[nt] = __builtin_amdgcn_wmma_f32_16x16x32_bf16(
          false, ap, false, bvf, (short)0, acc2[nt], false, false);
    }
  }
  float* soc = so + ((size_t)bh * NSORT + c * 64) * DHEAD;
#pragma unroll
  for (int nt = 0; nt < 6; nt++)
#pragma unroll
    for (int j2 = 0; j2 < 8; j2++)
      soc[(size_t)(rbase + j2) * DHEAD + nt * 16 + ncol] = acc2[nt][j2];
}

__global__ __launch_bounds__(256)
void lsh_combine_kernel(const float* __restrict__ so, const float* __restrict__ slog,
                        const int* __restrict__ undo, __bf16* __restrict__ obf) {
  int idx = blockIdx.x * 256 + threadIdx.x;  // NBH*SEQ
  int bh = idx >> 10, t = idx & (SEQ - 1);
  int sp[NHASH]; float lg[NHASH];
#pragma unroll
  for (int h = 0; h < NHASH; h++) {
    int i = h * SEQ + t;
    sp[h] = undo[(size_t)bh * NSORT + i];
    lg[h] = slog[(size_t)bh * NSORT + sp[h]];
  }
  float mx = lg[0];
#pragma unroll
  for (int h = 1; h < NHASH; h++) mx = fmaxf(mx, lg[h]);
  float w[NHASH]; float sum = 0.f;
#pragma unroll
  for (int h = 0; h < NHASH; h++) { w[h] = __expf(lg[h] - mx); sum += w[h]; }
  float inv = 1.f / sum;
  int b = bh >> 3, hh = bh & 7;
  __bf16* orow = obf + ((size_t)(b * SEQ + t)) * DIM + hh * DHEAD;
  for (int e = 0; e < DHEAD; e++) {
    float s = 0.f;
#pragma unroll
    for (int h = 0; h < NHASH; h++)
      s += w[h] * so[((size_t)bh * NSORT + sp[h]) * DHEAD + e];
    orow[e] = f2bf(s * inv);
  }
}

// ---------------------------------------------------------------------------
// Classifier head
// ---------------------------------------------------------------------------
__global__ __launch_bounds__(256)
void pool_kernel(const float* __restrict__ x1, const float* __restrict__ x2,
                 const float* __restrict__ pw, const float* __restrict__ pb,
                 float* __restrict__ pooled) {
  int idx = blockIdx.x * 256 + threadIdx.x;  // NBATCH*DIM
  int b = idx / DIM, n = idx % DIM;
  const float* r1 = x1 + (size_t)b * SEQ * DIM;  // token 0
  const float* r2 = x2 + (size_t)b * SEQ * DIM;
  float s = pb[n];
  for (int k = 0; k < DIM; k++) s += 0.5f * (r1[k] + r2[k]) * pw[(size_t)k * DIM + n];
  pooled[idx] = tanhf(s);
}

__global__ __launch_bounds__(256)
void logits_kernel(const float* __restrict__ pooled, const float* __restrict__ ow,
                   const float* __restrict__ ob, float* __restrict__ out) {
  int idx = threadIdx.x;
  if (idx < NBATCH * 10) {
    int b = idx / 10, cc = idx % 10;
    float s = ob[cc];
    for (int k = 0; k < DIM; k++) s += pooled[b * DIM + k] * ow[k * 10 + cc];
    out[idx] = s;
  }
}

// ---------------------------------------------------------------------------
extern "C" void kernel_launch(void* const* d_in, const int* in_sizes, int n_in,
                              void* d_out, int out_size, void* d_ws, size_t ws_size,
                              hipStream_t stream)
{
  const int*   x       = (const int*)  d_in[0];
  const float* tok_emb = (const float*)d_in[1];
  const float* pos_emb = (const float*)d_in[2];
  const float* ln1_g   = (const float*)d_in[3];
  const float* ln1_b   = (const float*)d_in[4];
  const float* Wqk     = (const float*)d_in[5];
  const float* Wv      = (const float*)d_in[6];
  const float* Wo      = (const float*)d_in[7];
  const float* bo      = (const float*)d_in[8];
  const float* ln2_g   = (const float*)d_in[9];
  const float* ln2_b   = (const float*)d_in[10];
  const float* W1      = (const float*)d_in[11];
  const float* b1      = (const float*)d_in[12];
  const float* W2      = (const float*)d_in[13];
  const float* b2      = (const float*)d_in[14];
  const float* pool_w  = (const float*)d_in[15];
  const float* pool_b  = (const float*)d_in[16];
  const float* out_w   = (const float*)d_in[17];
  const float* out_b   = (const float*)d_in[18];

  size_t off = 0;
  auto alloc = [&](size_t bytes) -> char* {
    char* p = (char*)d_ws + off;
    off += (bytes + 255) & ~(size_t)255;
    return p;
  };
  float*  x1      = (float*) alloc((size_t)NROWS * DIM * 4);
  float*  x2      = (float*) alloc((size_t)NROWS * DIM * 4);
  __bf16* abf     = (__bf16*)alloc((size_t)NROWS * DIM * 2);
  float*  qk      = (float*) alloc((size_t)NBH * SEQ * DHEAD * 4);
  float*  vvb     = (float*) alloc((size_t)NBH * SEQ * DHEAD * 4);
  __bf16* obf     = (__bf16*)alloc((size_t)NROWS * DIM * 2);
  __bf16* gbf     = (__bf16*)alloc((size_t)NROWS * FDIM * 2);
  float*  so      = (float*) alloc((size_t)NBH * NSORT * DHEAD * 4);
  float*  slog    = (float*) alloc((size_t)NBH * NSORT * 4);
  int*    buckets = (int*)   alloc((size_t)NBH * NSORT * 4);
  int*    sticker = (int*)   alloc((size_t)NBH * NSORT * 4);
  int*    undo    = (int*)   alloc((size_t)NBH * NSORT * 4);
  float*  rot     = (float*) alloc((size_t)DHEAD * 32 * 4);
  __bf16* WqkT    = (__bf16*)alloc((size_t)DEPTH * DIM * DIM * 2);   // [l][N][K]
  __bf16* WvT     = (__bf16*)alloc((size_t)DEPTH * DIM * DIM * 2);
  __bf16* WoT     = (__bf16*)alloc((size_t)DEPTH * DIM * DIM * 2);
  __bf16* W1T     = (__bf16*)alloc((size_t)DEPTH * DIM * FDIM * 2);  // [l][FDIM][DIM]
  __bf16* W2T     = (__bf16*)alloc((size_t)DEPTH * DIM * FDIM * 2);  // [l][DIM][FDIM]
  float*  pooled  = (float*) alloc((size_t)NBATCH * DIM * 4);

  dim3 tb(32, 8);
  transpose_bf16_kernel<<<dim3(DIM / 32,  DIM / 32,  DEPTH), tb, 0, stream>>>(Wqk, WqkT, DIM,  DIM);
  transpose_bf16_kernel<<<dim3(DIM / 32,  DIM / 32,  DEPTH), tb, 0, stream>>>(Wv,  WvT,  DIM,  DIM);
  transpose_bf16_kernel<<<dim3(DIM / 32,  DIM / 32,  DEPTH), tb, 0, stream>>>(Wo,  WoT,  DIM,  DIM);
  transpose_bf16_kernel<<<dim3(DIM / 32,  FDIM / 32, DEPTH), tb, 0, stream>>>(W1,  W1T,  DIM,  FDIM);
  transpose_bf16_kernel<<<dim3(FDIM / 32, DIM / 32,  DEPTH), tb, 0, stream>>>(W2,  W2T,  FDIM, DIM);

  embed_kernel<<<(NROWS * DIM) / 256, 256, 0, stream>>>(x, tok_emb, pos_emb, x1, x2);

  for (int i = 0; i < DEPTH; i++) {
    layernorm_bf16_kernel<<<NROWS, 256, 0, stream>>>(x2, ln1_g + i * DIM, ln1_b + i * DIM, abf);
    gemm_bf16_wmma_kernel<<<dim3(NROWS / GM, DIM / GN), 256, 0, stream>>>(
        abf, WqkT + (size_t)i * DIM * DIM, nullptr, nullptr, qk, NROWS, DIM, DIM, 1);
    gemm_bf16_wmma_kernel<<<dim3(NROWS / GM, DIM / GN), 256, 0, stream>>>(
        abf, WvT + (size_t)i * DIM * DIM, nullptr, nullptr, vvb, NROWS, DIM, DIM, 1);

    rotgen_kernel<<<(DHEAD * 32 + 255) / 256, 256, 0, stream>>>(rot, i);
    lsh_hash_kernel<<<(NBH * SEQ) / 256, 256, 0, stream>>>(qk, rot, buckets);
    lsh_sort_kernel<<<NBH, 1024, 0, stream>>>(buckets, sticker, undo);
    lsh_attn_kernel<<<dim3(NCHUNK, NBH), 128, 0, stream>>>(qk, vvb, sticker, so, slog);
    lsh_combine_kernel<<<(NBH * SEQ) / 256, 256, 0, stream>>>(so, slog, undo, obf);

    gemm_bf16_wmma_kernel<<<dim3(NROWS / GM, DIM / GN), 256, 0, stream>>>(
        obf, WoT + (size_t)i * DIM * DIM, bo + i * DIM, x1, x1, NROWS, DIM, DIM, 2);

    layernorm_bf16_kernel<<<NROWS, 256, 0, stream>>>(x1, ln2_g + i * DIM, ln2_b + i * DIM, abf);
    gemm_bf16_wmma_kernel<<<dim3(NROWS / GM, FDIM / GN), 256, 0, stream>>>(
        abf, W1T + (size_t)i * DIM * FDIM, b1 + i * FDIM, nullptr, gbf, NROWS, FDIM, DIM, 3);
    gemm_bf16_wmma_kernel<<<dim3(NROWS / GM, DIM / GN), 256, 0, stream>>>(
        gbf, W2T + (size_t)i * DIM * FDIM, b2 + i * DIM, x2, x2, NROWS, DIM, FDIM, 2);
  }

  pool_kernel<<<(NBATCH * DIM) / 256, 256, 0, stream>>>(x1, x2, pool_w, pool_b, pooled);
  logits_kernel<<<1, 256, 0, stream>>>(pooled, out_w, out_b, (float*)d_out);

  (void)in_sizes; (void)n_in; (void)out_size; (void)ws_size;
}
